// TKL_aio_1_42674795053925
// MI455X (gfx1250) — compile-verified, matrix operand order checked by reading, not run on hardware
//
#include <hip/hip_runtime.h>
#include <math.h>

// ---------------- model constants (from reference) ----------------
#define BQ 16
#define QLEN 30
#define DMOD 256
#define HNUM 8
#define HD 32
#define FFD 1024
#define LNUM 2
#define CHUNK 40
#define OVERLAP 5
#define EXTL 50
#define SWIN 30
#define NKER 11
#define SLABC 160                 // chunks per slab (8000 rows -> L2 resident)
#define SLABROWS (SLABC * EXTL)

typedef __attribute__((ext_vector_type(16))) __bf16 v16bf;
typedef __attribute__((ext_vector_type(8)))  float  v8f;

__device__ __forceinline__ unsigned short f2b(float f) {
  unsigned x = __float_as_uint(f);
  return (unsigned short)((x + 0x7fffu + ((x >> 16) & 1u)) >> 16);  // RNE
}

// Load one wave-operand tile (16x32 bf16) per the CDNA5 16-bit A-layout.
// p16 must already point at: base + row*K + k0 + half*8 (bf16 units).
__device__ __forceinline__ v16bf load_tile16(const unsigned short* p16) {
  const unsigned* p = (const unsigned*)p16;
  union { v16bf v; unsigned u[8]; } t;
#pragma unroll
  for (int i = 0; i < 4; ++i) { t.u[i] = p[i]; t.u[4 + i] = p[8 + i]; }
  return t.v;
}

__device__ __forceinline__ v8f wmma_bf16(v16bf a, v16bf b, v8f c) {
  return __builtin_amdgcn_wmma_f32_16x16x32_bf16(false, a, false, b, (short)0, c,
                                                 false, false);
}

// ---------------- generic bf16 WMMA GEMM: C = A(MxK) * W(NxK)^T + bias ------
// optional residual add (f32 MxN), relu, f32 and/or bf16 store.
__global__ void __launch_bounds__(256)
gemm_bf16_wmma(const unsigned short* __restrict__ A,
               const unsigned short* __restrict__ W,
               const float* __restrict__ bias, const float* __restrict__ resid,
               float* __restrict__ Cf, unsigned short* __restrict__ Cb,
               int M, int N, int K, int relu) {
  int tiles_n = N >> 4;
  int tiles_m = (M + 15) >> 4;
  int wave = (blockIdx.x << 3) + (threadIdx.x >> 5);
  if (wave >= tiles_m * tiles_n) return;          // wave-uniform: EXEC stays full
  int mi = wave / tiles_n, ni = wave % tiles_n;
  int lane = threadIdx.x & 31;
  int half = lane >> 4, lr = lane & 15;
  int ar = mi * 16 + lr; if (ar >= M) ar = M - 1; // clamp (stores are guarded)
  int br = ni * 16 + lr;
  const unsigned short* Ap = A + (size_t)ar * K + half * 8;
  const unsigned short* Wp = W + (size_t)br * K + half * 8;
  v8f acc = {0.f, 0.f, 0.f, 0.f, 0.f, 0.f, 0.f, 0.f};
  for (int k0 = 0; k0 < K; k0 += 32) {
    v16bf a = load_tile16(Ap + k0);
    v16bf b = load_tile16(Wp + k0);
    acc = wmma_bf16(a, b, acc);
  }
  int col = ni * 16 + lr;
  float bv = bias ? bias[col] : 0.f;
#pragma unroll
  for (int r8 = 0; r8 < 8; ++r8) {
    int row = mi * 16 + r8 + half * 8;
    if (row < M) {
      float v = acc[r8] + bv;
      if (resid) v += resid[(size_t)row * N + col];
      if (relu && v < 0.f) v = 0.f;
      if (Cf) Cf[(size_t)row * N + col] = v;
      if (Cb) Cb[(size_t)row * N + col] = f2b(v);
    }
  }
}

// ---------------- cosine einsum via WMMA: per chunk (30x40) = qn(30,256)@dn(40,256)^T
__global__ void __launch_bounds__(192)
cos_wmma_kernel(const unsigned short* __restrict__ qn,
                const unsigned short* __restrict__ dn,
                float* __restrict__ cosout, int n_chunks) {
  int c = blockIdx.x;
  int b = c / n_chunks;
  int wave = threadIdx.x >> 5;      // 6 waves: 2 m-tiles x 3 n-tiles
  int mi = wave / 3, ni = wave % 3;
  int lane = threadIdx.x & 31, half = lane >> 4, lr = lane & 15;
  int qr = mi * 16 + lr; if (qr >= QLEN) qr = QLEN - 1;
  int dr = ni * 16 + lr; if (dr >= CHUNK) dr = CHUNK - 1;
  const unsigned short* qp = qn + ((size_t)b * QLEN + qr) * DMOD + half * 8;
  const unsigned short* dp = dn + ((size_t)c * CHUNK + dr) * DMOD + half * 8;
  v8f acc = {0.f, 0.f, 0.f, 0.f, 0.f, 0.f, 0.f, 0.f};
  for (int k0 = 0; k0 < DMOD; k0 += 32) {
    v16bf a = load_tile16(qp + k0);
    v16bf bb = load_tile16(dp + k0);
    acc = wmma_bf16(a, bb, acc);
  }
  int col = ni * 16 + lr;
#pragma unroll
  for (int r8 = 0; r8 < 8; ++r8) {
    int row = mi * 16 + r8 + half * 8;
    if (row < QLEN && col < CHUNK)
      cosout[((size_t)c * QLEN + row) * CHUNK + col] = acc[r8];
  }
}

// ---------------- elementwise helpers ----------------
__global__ void f2b_kernel(const float* __restrict__ in,
                           unsigned short* __restrict__ out, long n) {
  long i = (long)blockIdx.x * blockDim.x + threadIdx.x;
  if (i < n) out[i] = f2b(in[i]);
}

__global__ void zero_kernel(float* p, int n) {
  int i = blockIdx.x * blockDim.x + threadIdx.x;
  if (i < n) p[i] = 0.f;
}

// ---------------- embedding / chunking ----------------
__global__ void embed_q_kernel(const int* __restrict__ qt,
                               const float* __restrict__ emb,
                               const float* __restrict__ pos,
                               float* __restrict__ x0, float* __restrict__ qmask) {
  int idx = blockIdx.x * blockDim.x + threadIdx.x;
  if (idx >= BQ * QLEN * DMOD) return;
  int row = idx >> 8, d = idx & 255;
  int tok = qt[row];
  x0[idx] = emb[(size_t)tok * DMOD + d] + pos[(row % QLEN) * DMOD + d];
  if (d == 0) qmask[row] = (tok > 0) ? 1.f : 0.f;
}

__global__ void build_chunks_kernel(const int* __restrict__ doc,
                                    int* __restrict__ cid,
                                    float* __restrict__ dmask, int n_chunks,
                                    int dlen) {
  int idx = blockIdx.x * blockDim.x + threadIdx.x;
  int tot = BQ * n_chunks * EXTL;
  if (idx >= tot) return;
  int rc = idx / EXTL, e = idx % EXTL;
  int b = rc / n_chunks, c = rc % n_chunks;
  int src = c * CHUNK + e - OVERLAP;
  int id = (src >= 0 && src < dlen) ? doc[(size_t)b * dlen + src] : 0;
  cid[idx] = id;
  dmask[idx] = (id > 0) ? 1.f : 0.f;
}

__global__ void packed_kernel(const int* __restrict__ cid,
                              float* __restrict__ packed, int NC) {
  int rc = blockIdx.x * blockDim.x + threadIdx.x;
  if (rc >= NC) return;
  long s = 0;
  for (int e = OVERLAP; e < EXTL - OVERLAP; ++e) s += cid[(size_t)rc * EXTL + e];
  packed[rc] = (s != 0) ? 1.f : 0.f;
}

__global__ void embed_d_kernel(const int* __restrict__ cid_slab,
                               const float* __restrict__ emb,
                               const float* __restrict__ posd,
                               float* __restrict__ x0, int rows) {
  long idx = (long)blockIdx.x * blockDim.x + threadIdx.x;
  if (idx >= (long)rows * DMOD) return;
  int r = (int)(idx >> 8), d = (int)(idx & 255);
  int e = r % EXTL;
  int tok = cid_slab[r];
  x0[idx] = emb[(size_t)tok * DMOD + d] + posd[e * DMOD + d];
}

// ---------------- attention (one block per sequence x head) ----------------
__global__ void __launch_bounds__(64)
attn_kernel(const float* __restrict__ qkv, unsigned short* __restrict__ obf,
            const float* __restrict__ mask, int S) {
  __shared__ float kk[64 * HD];
  __shared__ float vv[64 * HD];
  __shared__ float ss[64 * 64];
  int seq = blockIdx.x, h = blockIdx.y;
  int tid = threadIdx.x;
  const float* base = qkv + (size_t)seq * S * (3 * DMOD);
  for (int i = tid; i < S; i += 64) {
    const float* kr = base + (size_t)i * 3 * DMOD + DMOD + h * HD;
    const float* vr = base + (size_t)i * 3 * DMOD + 2 * DMOD + h * HD;
    for (int d = 0; d < HD; ++d) { kk[i * HD + d] = kr[d]; vv[i * HD + d] = vr[d]; }
  }
  __syncthreads();
  if (tid < S) {
    float qreg[HD];
    const float* qr = base + (size_t)tid * 3 * DMOD + h * HD;
#pragma unroll
    for (int d = 0; d < HD; ++d) qreg[d] = qr[d];
    float mx = -1e30f;
    for (int j = 0; j < S; ++j) {
      float dot = 0.f;
#pragma unroll
      for (int d = 0; d < HD; ++d) dot += qreg[d] * kk[j * HD + d];
      dot *= 0.17677669529663688f;                  // 1/sqrt(32)
      dot = (mask[seq * S + j] > 0.f) ? dot : -1e9f;
      ss[tid * 64 + j] = dot;
      if (dot > mx) mx = dot;
    }
    float sum = 0.f;
    for (int j = 0; j < S; ++j) {
      float e = __expf(ss[tid * 64 + j] - mx);
      ss[tid * 64 + j] = e;
      sum += e;
    }
    float inv = 1.f / sum;
    for (int d = 0; d < HD; ++d) {
      float o = 0.f;
      for (int j = 0; j < S; ++j) o += ss[tid * 64 + j] * vv[j * HD + d];
      obf[((size_t)seq * S + tid) * DMOD + h * HD + d] = f2b(o * inv);
    }
  }
}

// ---------------- layernorm (block per row, D=256) ----------------
__global__ void __launch_bounds__(256)
ln_kernel(const float* __restrict__ xin, float* __restrict__ xout,
          const float* __restrict__ g, const float* __restrict__ b) {
  __shared__ float red[256];
  int row = blockIdx.x, t = threadIdx.x;
  float v = xin[(size_t)row * DMOD + t];
  red[t] = v;
  __syncthreads();
  for (int s = 128; s > 0; s >>= 1) { if (t < s) red[t] += red[t + s]; __syncthreads(); }
  float mean = red[0] * (1.f / DMOD);
  __syncthreads();
  float d = v - mean;
  red[t] = d * d;
  __syncthreads();
  for (int s = 128; s > 0; s >>= 1) { if (t < s) red[t] += red[t + s]; __syncthreads(); }
  float var = red[0] * (1.f / DMOD);
  xout[(size_t)row * DMOD + t] = d * rsqrtf(var + 1e-5f) * g[t] + b[t];
}

// ---------------- mix + normalize (writes normalized bf16 for cosine) -------
__global__ void __launch_bounds__(256)
mix_q_kernel(const float* __restrict__ x0, const float* __restrict__ x,
             const float* __restrict__ qmask, const float* __restrict__ mixer,
             unsigned short* __restrict__ qn) {
  __shared__ float red[256];
  int row = blockIdx.x, t = threadIdx.x;
  float mx = *mixer;
  float v = (mx * x0[(size_t)row * DMOD + t] + (1.f - mx) * x[(size_t)row * DMOD + t]) *
            qmask[row];
  red[t] = v * v;
  __syncthreads();
  for (int s = 128; s > 0; s >>= 1) { if (t < s) red[t] += red[t + s]; __syncthreads(); }
  float nrm = sqrtf(fmaxf(red[0], 1e-26f));
  qn[(size_t)row * DMOD + t] = f2b(v / nrm);
}

__global__ void __launch_bounds__(256)
mix_d_kernel(const float* __restrict__ x0, const float* __restrict__ x,
             const float* __restrict__ dmask_slab, const float* __restrict__ mixer,
             unsigned short* __restrict__ dn, int cs0) {
  __shared__ float red[256];
  int lm = blockIdx.x, t = threadIdx.x;
  int lc = lm / CHUNK, p = lm % CHUNK;
  int lr = lc * EXTL + OVERLAP + p;
  float mx = *mixer;
  float v = (mx * x0[(size_t)lr * DMOD + t] + (1.f - mx) * x[(size_t)lr * DMOD + t]) *
            dmask_slab[lr];
  red[t] = v * v;
  __syncthreads();
  for (int s = 128; s > 0; s >>= 1) { if (t < s) red[t] += red[t + s]; __syncthreads(); }
  float nrm = sqrtf(fmaxf(red[0], 1e-26f));
  dn[((size_t)(cs0 + lc) * CHUNK + p) * DMOD + t] = f2b(v / nrm);
}

// ---------------- kernel pooling: one block per (b,q); LDS-resident row -----
__global__ void __launch_bounds__(256)
kpool_kernel(const float* __restrict__ cosb, const float* __restrict__ dmask,
             const float* __restrict__ packed, const float* __restrict__ qmask,
             const float* __restrict__ mu, const float* __restrict__ sigma,
             const float* __restrict__ dw, float* __restrict__ score,
             int n_chunks, int nw) {
  extern __shared__ float kb[];  // TOT * 12 floats (11 kernels + indicator)
  int b = blockIdx.x / QLEN, q = blockIdx.x % QLEN;
  int TOT = n_chunks * CHUNK;
  for (int t = threadIdx.x; t < TOT; t += 256) {
    int ci = t / CHUNK, pos = t % CHUNK;
    int c = b * n_chunks + ci;
    float cv = cosb[((size_t)c * QLEN + q) * CHUNK + pos];
    float m = dmask[(size_t)c * EXTL + OVERLAP + pos] * packed[c];
    float s = 0.f;
#pragma unroll
    for (int k = 0; k < NKER; ++k) {
      float d = cv - mu[k];
      float sg = sigma[k];
      float val = __expf(-d * d / (2.f * sg * sg)) * m;
      kb[t * 12 + k] = val;
      s += val;
    }
    kb[t * 12 + NKER] = (s != 0.f) ? 1.f : 0.f;
  }
  __syncthreads();
  float qm = qmask[b * QLEN + q];
  for (int w = threadIdx.x; w < nw; w += 256) {
    int st = 2 * w;
    float ws[NKER];
#pragma unroll
    for (int k = 0; k < NKER; ++k) ws[k] = 0.f;
    float wl = 0.f;
#pragma unroll
    for (int i = 0; i < SWIN; ++i) {
      int t = st + i;
#pragma unroll
      for (int k = 0; k < NKER; ++k) ws[k] += kb[t * 12 + k];
      wl += kb[t * 12 + NKER];
    }
    if (wl > 0.f) {
      float p = 0.f;
#pragma unroll
      for (int k = 0; k < NKER; ++k) p += __logf(fmaxf(ws[k], 1e-10f)) * dw[k];
      atomicAdd(&score[(size_t)b * nw + w], p * qm);
    }
  }
}

// ---------------- top-k + neighborhood scoring ----------------
__global__ void final_kernel(float* __restrict__ score, float* __restrict__ msc,
                             const float* __restrict__ cscore,
                             float* __restrict__ out, int nw) {
  int b = threadIdx.x;
  if (b >= BQ) return;
  float* sc = score + (size_t)b * nw;
  float* ms = msc + (size_t)b * nw;
  for (int w = 0; w < nw; ++w) {
    float s = sc[w];
    if (s == 0.f) s = -9900.f;
    sc[w] = s;
    ms[w] = s;
  }
  int tops[3];
  for (int c = 0; c < 3; ++c) {
    int best = 0;
    float bv = ms[0];
    for (int w = 1; w < nw; ++w)
      if (ms[w] > bv) { bv = ms[w]; best = w; }
    tops[c] = best;
    for (int w = 0; w < nw; ++w) {
      int d = w - best; if (d < 0) d = -d;
      if (d < SWIN / 2) ms[w] = -10001.f - (float)c;
    }
  }
  const int offs[5] = {0, -1, -2, 1, 2};
  float sum = 0.f;
  for (int o5 = 0; o5 < 5; ++o5)
    for (int j = 0; j < 3; ++j) {
      int idx = tops[j] + offs[o5];
      if (idx < 0) idx = 0;
      if (idx > nw - 1) idx = nw - 1;
      float g = sc[idx];
      if (g <= -9900.f) g = 0.f;
      sum += g * cscore[o5 * 3 + j];
    }
  out[b] = sum;
}

// =====================================================================
extern "C" void kernel_launch(void* const* d_in, const int* in_sizes, int n_in,
                              void* d_out, int out_size, void* d_ws, size_t ws_size,
                              hipStream_t stream) {
  const int* qt      = (const int*)d_in[0];
  const int* doc     = (const int*)d_in[1];
  const float* emb   = (const float*)d_in[4];
  const float* pos_q = (const float*)d_in[5];
  const float* pos_d = (const float*)d_in[6];
  const float* mixer = (const float*)d_in[7];
  const float* mu    = (const float*)d_in[8];
  const float* sigma = (const float*)d_in[9];
  const float* qkv_w = (const float*)d_in[10];
  const float* qkv_b = (const float*)d_in[11];
  const float* out_w = (const float*)d_in[12];
  const float* out_b = (const float*)d_in[13];
  const float* ln1g  = (const float*)d_in[14];
  const float* ln1b  = (const float*)d_in[15];
  const float* ff1w  = (const float*)d_in[16];
  const float* ff1b  = (const float*)d_in[17];
  const float* ff2w  = (const float*)d_in[18];
  const float* ff2b  = (const float*)d_in[19];
  const float* ln2g  = (const float*)d_in[20];
  const float* ln2b  = (const float*)d_in[21];
  const float* dw    = (const float*)d_in[22];
  const float* csc   = (const float*)d_in[23];

  int dlen = in_sizes[1] / BQ;
  int pad_r = (dlen > OVERLAP) ? (EXTL - (dlen - OVERLAP) % CHUNK)
                               : (EXTL - OVERLAP - dlen);
  int padded = dlen + OVERLAP + pad_r;
  int n_chunks = (padded - EXTL) / CHUNK + 1;
  int NC = BQ * n_chunks;
  int TOT = n_chunks * CHUNK;
  int nw = (TOT - SWIN) / 2 + 1;
  int NQ = BQ * QLEN;

  // ---- carve workspace (each buffer 256B-aligned) ----
  char* base = (char*)d_ws;
  size_t off = 0;
  auto alloc = [&](size_t bytes) -> void* {
    void* p = base + off;
    off = (off + bytes + 255) & ~(size_t)255;
    return p;
  };
  unsigned short* w_qkv_bf = (unsigned short*)alloc((size_t)LNUM * 3 * DMOD * DMOD * 2);
  unsigned short* w_out_bf = (unsigned short*)alloc((size_t)LNUM * DMOD * DMOD * 2);
  unsigned short* w_ff1_bf = (unsigned short*)alloc((size_t)LNUM * FFD * DMOD * 2);
  unsigned short* w_ff2_bf = (unsigned short*)alloc((size_t)LNUM * DMOD * FFD * 2);
  int*   chunk_ids = (int*)alloc((size_t)NC * EXTL * 4);
  float* dmask     = (float*)alloc((size_t)NC * EXTL * 4);
  float* packed    = (float*)alloc((size_t)NC * 4);
  float* qmask     = (float*)alloc((size_t)NQ * 4);
  float* q_x0  = (float*)alloc((size_t)NQ * DMOD * 4);
  float* q_x   = (float*)alloc((size_t)NQ * DMOD * 4);
  float* q_tmp = (float*)alloc((size_t)NQ * DMOD * 4);
  float* q_qkv = (float*)alloc((size_t)NQ * 3 * DMOD * 4);
  float* d_x0  = (float*)alloc((size_t)SLABROWS * DMOD * 4);
  float* d_x   = (float*)alloc((size_t)SLABROWS * DMOD * 4);
  float* d_tmp = (float*)alloc((size_t)SLABROWS * DMOD * 4);
  float* d_qkv = (float*)alloc((size_t)SLABROWS * 3 * DMOD * 4);
  unsigned short* act_bf = (unsigned short*)alloc((size_t)SLABROWS * DMOD * 2);
  unsigned short* ffh_bf = (unsigned short*)alloc((size_t)SLABROWS * FFD * 2);
  unsigned short* qn_bf  = (unsigned short*)alloc((size_t)NQ * DMOD * 2);
  unsigned short* dn_bf  = (unsigned short*)alloc((size_t)NC * CHUNK * DMOD * 2);
  float* cosbuf = (float*)alloc((size_t)NC * QLEN * CHUNK * 4);
  float* score  = (float*)alloc((size_t)BQ * nw * 4);
  float* mscbuf = (float*)alloc((size_t)BQ * nw * 4);

  auto cvt = [&](const float* src, unsigned short* dst, long n) {
    f2b_kernel<<<(int)((n + 255) / 256), 256, 0, stream>>>(src, dst, n);
  };
  auto gemm = [&](const unsigned short* A, const unsigned short* W,
                  const float* bias, const float* resid, float* Cf,
                  unsigned short* Cb, int M, int N, int K, int relu) {
    int tiles = ((M + 15) / 16) * (N / 16);
    gemm_bf16_wmma<<<(tiles + 7) / 8, 256, 0, stream>>>(A, W, bias, resid, Cf, Cb,
                                                        M, N, K, relu);
  };

  // ---- one-time bf16 weight conversion ----
  cvt(qkv_w, w_qkv_bf, (long)LNUM * 3 * DMOD * DMOD);
  cvt(out_w, w_out_bf, (long)LNUM * DMOD * DMOD);
  cvt(ff1w,  w_ff1_bf, (long)LNUM * FFD * DMOD);
  cvt(ff2w,  w_ff2_bf, (long)LNUM * DMOD * FFD);

  // ---- query side ----
  embed_q_kernel<<<(NQ * DMOD + 255) / 256, 256, 0, stream>>>(qt, emb, pos_q, q_x0,
                                                              qmask);
  {
    const float* cur = q_x0;
    for (int l = 0; l < LNUM; ++l) {
      cvt(cur, act_bf, (long)NQ * DMOD);
      gemm(act_bf, w_qkv_bf + (size_t)l * 3 * DMOD * DMOD, qkv_b + l * 3 * DMOD,
           nullptr, q_qkv, nullptr, NQ, 3 * DMOD, DMOD, 0);
      attn_kernel<<<dim3(BQ, HNUM), 64, 0, stream>>>(q_qkv, act_bf, qmask, QLEN);
      gemm(act_bf, w_out_bf + (size_t)l * DMOD * DMOD, out_b + l * DMOD, cur, q_tmp,
           nullptr, NQ, DMOD, DMOD, 0);
      ln_kernel<<<NQ, 256, 0, stream>>>(q_tmp, q_x, ln1g + l * DMOD, ln1b + l * DMOD);
      cvt(q_x, act_bf, (long)NQ * DMOD);
      gemm(act_bf, w_ff1_bf + (size_t)l * FFD * DMOD, ff1b + l * FFD, nullptr,
           nullptr, ffh_bf, NQ, FFD, DMOD, 1);
      gemm(ffh_bf, w_ff2_bf + (size_t)l * DMOD * FFD, ff2b + l * DMOD, q_x, q_tmp,
           nullptr, NQ, DMOD, FFD, 0);
      ln_kernel<<<NQ, 256, 0, stream>>>(q_tmp, q_x, ln2g + l * DMOD, ln2b + l * DMOD);
      cur = q_x;
    }
  }
  mix_q_kernel<<<NQ, 256, 0, stream>>>(q_x0, q_x, qmask, mixer, qn_bf);

  // ---- document side: chunking, then L2-resident slabs ----
  build_chunks_kernel<<<(NC * EXTL + 255) / 256, 256, 0, stream>>>(doc, chunk_ids,
                                                                   dmask, n_chunks,
                                                                   dlen);
  packed_kernel<<<(NC + 255) / 256, 256, 0, stream>>>(chunk_ids, packed, NC);

  for (int cs0 = 0; cs0 < NC; cs0 += SLABC) {
    int cn = (NC - cs0 < SLABC) ? (NC - cs0) : SLABC;
    int rows = cn * EXTL;
    embed_d_kernel<<<(int)(((long)rows * DMOD + 255) / 256), 256, 0, stream>>>(
        chunk_ids + (size_t)cs0 * EXTL, emb, pos_d, d_x0, rows);
    const float* cur = d_x0;
    for (int l = 0; l < LNUM; ++l) {
      cvt(cur, act_bf, (long)rows * DMOD);
      gemm(act_bf, w_qkv_bf + (size_t)l * 3 * DMOD * DMOD, qkv_b + l * 3 * DMOD,
           nullptr, d_qkv, nullptr, rows, 3 * DMOD, DMOD, 0);
      attn_kernel<<<dim3(cn, HNUM), 64, 0, stream>>>(d_qkv, act_bf,
                                                     dmask + (size_t)cs0 * EXTL,
                                                     EXTL);
      gemm(act_bf, w_out_bf + (size_t)l * DMOD * DMOD, out_b + l * DMOD, cur, d_tmp,
           nullptr, rows, DMOD, DMOD, 0);
      ln_kernel<<<rows, 256, 0, stream>>>(d_tmp, d_x, ln1g + l * DMOD,
                                          ln1b + l * DMOD);
      cvt(d_x, act_bf, (long)rows * DMOD);
      gemm(act_bf, w_ff1_bf + (size_t)l * FFD * DMOD, ff1b + l * FFD, nullptr,
           nullptr, ffh_bf, rows, FFD, DMOD, 1);
      gemm(ffh_bf, w_ff2_bf + (size_t)l * DMOD * FFD, ff2b + l * DMOD, d_x, d_tmp,
           nullptr, rows, DMOD, FFD, 0);
      ln_kernel<<<rows, 256, 0, stream>>>(d_tmp, d_x, ln2g + l * DMOD,
                                          ln2b + l * DMOD);
      cur = d_x;
    }
    mix_d_kernel<<<cn * CHUNK, 256, 0, stream>>>(d_x0, d_x,
                                                 dmask + (size_t)cs0 * EXTL, mixer,
                                                 dn_bf, cs0);
  }

  // ---- cosine (WMMA) + kernel pooling + final scoring ----
  cos_wmma_kernel<<<NC, 192, 0, stream>>>(qn_bf, dn_bf, cosbuf, n_chunks);
  zero_kernel<<<(BQ * nw + 255) / 256, 256, 0, stream>>>(score, BQ * nw);
  kpool_kernel<<<NQ, 256, (size_t)TOT * 12 * sizeof(float), stream>>>(
      cosbuf, dmask, packed, qmask, mu, sigma, dw, score, n_chunks, nw);
  final_kernel<<<1, 32, 0, stream>>>(score, mscbuf, csc, (float*)d_out, nw);
}
